// Model_28544352649821
// MI455X (gfx1250) — matrix operand order repaired and checked by reading, not compiled, not run on hardware
//
#include <hip/hip_runtime.h>
#include <math.h>

// ---------------------------------------------------------------------------
// Types for CDNA5 WMMA (gfx1250, wave32)
// ---------------------------------------------------------------------------
typedef __bf16 bf16;
typedef __bf16 v16bf __attribute__((ext_vector_type(16)));
typedef __bf16 v8bf  __attribute__((ext_vector_type(8)));
typedef float  v8f   __attribute__((ext_vector_type(8)));

// ---------------------------------------------------------------------------
// Model constants
// ---------------------------------------------------------------------------
constexpr int  kB    = 256;
constexpr int  kT    = 27;
constexpr int  kS    = 17;
constexpr int  kC    = 256;
constexpr int  kHALF = 128;
constexpr long kNT   = (long)kB * kT * kS;   // 117504 tokens (divisible by 32)
constexpr int  kROWT = (int)(kNT / 32);       // 3672 row tiles

__device__ __forceinline__ float gelu_exact(float x) {
    return 0.5f * x * (1.0f + erff(x * 0.70710678118654752f));
}

__device__ __forceinline__ v8f wmma_bf16(v16bf a, v16bf b, v8f c) {
    // D = A(16x32 bf16) * B(32x16 bf16) + C(16x16 f32)
    return __builtin_amdgcn_wmma_f32_16x16x32_bf16(
        false, a, false, b, (short)0, c, false, false);
}

// ---------------------------------------------------------------------------
// Generic WMMA GEMM:  Y[NT, Mout] = A[NT, K](bf16) @ W[Mout, K]^T(bf16) (+bias)
// (+res) (ACT==1 -> exact GELU). One wave computes a 32x64 output tile.
// K, Mout multiples of 32 / 64; rows multiple of 32.
// ---------------------------------------------------------------------------
template<int ACT, bool RESID, bool OUT_BF16>
__global__ __launch_bounds__(256) void gemm_wmma_kernel(
    const bf16* __restrict__ A, const bf16* __restrict__ W,
    const float* __restrict__ bias, const float* __restrict__ res,
    void* __restrict__ outp, int K, int Mout, int rowTiles)
{
    int wave = blockIdx.x * 8 + (threadIdx.x >> 5);
    int lane = threadIdx.x & 31;
    int colTiles = Mout >> 6;
    int rowTile  = wave / colTiles;
    if (rowTile >= rowTiles) return;
    int colTile = wave - rowTile * colTiles;
    long m0 = (long)rowTile * 32;
    int  n0 = colTile * 64;
    int  half = lane >> 4;       // 0/1
    int  nr   = lane & 15;       // A row-in-tile on load; C column on store

    v8f acc[2][4];
#pragma unroll
    for (int j = 0; j < 4; ++j) {
        float bv = bias ? bias[n0 + 16 * j + nr] : 0.0f;
        v8f z = {bv, bv, bv, bv, bv, bv, bv, bv};
        acc[0][j] = z; acc[1][j] = z;
    }

    const bf16* arow0 = A + (m0 + nr) * (long)K;
    const bf16* arow1 = A + (m0 + 16 + nr) * (long)K;

    for (int k = 0; k < K; k += 32) {
        v8bf lo0 = *(const v8bf*)(arow0 + k + half * 16);
        v8bf hi0 = *(const v8bf*)(arow0 + k + half * 16 + 8);
        v8bf lo1 = *(const v8bf*)(arow1 + k + half * 16);
        v8bf hi1 = *(const v8bf*)(arow1 + k + half * 16 + 8);
        v16bf a0 = __builtin_shufflevector(lo0, hi0, 0,1,2,3,4,5,6,7,8,9,10,11,12,13,14,15);
        v16bf a1 = __builtin_shufflevector(lo1, hi1, 0,1,2,3,4,5,6,7,8,9,10,11,12,13,14,15);
#pragma unroll
        for (int j = 0; j < 4; ++j) {
            v16bf bfrag = *(const v16bf*)(W + (long)(n0 + 16 * j + nr) * K + k + half * 16);
            acc[0][j] = wmma_bf16(a0, bfrag, acc[0][j]);
            acc[1][j] = wmma_bf16(a1, bfrag, acc[1][j]);
        }
    }

#pragma unroll
    for (int t = 0; t < 2; ++t) {
#pragma unroll
        for (int j = 0; j < 4; ++j) {
#pragma unroll
            for (int r = 0; r < 8; ++r) {
                long row = m0 + 16 * t + half * 8 + r;
                long idx = row * Mout + n0 + 16 * j + nr;
                float x = acc[t][j][r];
                if (RESID)   x += res[idx];
                if (ACT == 1) x = gelu_exact(x);
                if (OUT_BF16) ((bf16*)outp)[idx] = (bf16)x;
                else          ((float*)outp)[idx] = x;
            }
        }
    }
}

// ---------------------------------------------------------------------------
// f32 -> bf16 conversion
// ---------------------------------------------------------------------------
__global__ __launch_bounds__(256) void f32_to_bf16_kernel(
    const float* __restrict__ src, bf16* __restrict__ dst, long n)
{
    long i = (long)blockIdx.x * 256 + threadIdx.x;
    if (i < n) dst[i] = (bf16)src[i];
}

// ---------------------------------------------------------------------------
// Embedding: h = gelu(x @ emb_w^T), x:(NT,2), emb_w:(256,2)
// ---------------------------------------------------------------------------
__global__ __launch_bounds__(256) void embed_kernel(
    const float* __restrict__ x, const float* __restrict__ ew,
    float* __restrict__ h)
{
    long idx = (long)blockIdx.x * 256 + threadIdx.x;  // NT*256 threads
    long tok = idx >> 8; int c = (int)(idx & 255);
    float v = x[tok * 2] * ew[c * 2] + x[tok * 2 + 1] * ew[c * 2 + 1];
    h[idx] = gelu_exact(v);
}

// ---------------------------------------------------------------------------
// LayerNorm over C=256, one wave per token, bf16 output
// ---------------------------------------------------------------------------
__global__ __launch_bounds__(256) void layernorm_kernel(
    const float* __restrict__ x, const float* __restrict__ g,
    const float* __restrict__ b, bf16* __restrict__ out)
{
    long tok = (long)blockIdx.x * 8 + (threadIdx.x >> 5);
    int lane = threadIdx.x & 31;
    long base = tok * 256;
    float v[8];
    float s = 0.0f;
#pragma unroll
    for (int j = 0; j < 8; ++j) { v[j] = x[base + lane + 32 * j]; s += v[j]; }
#pragma unroll
    for (int off = 16; off >= 1; off >>= 1) s += __shfl_xor(s, off, 32);
    float mean = s * (1.0f / 256.0f);
    float q = 0.0f;
#pragma unroll
    for (int j = 0; j < 8; ++j) { float d = v[j] - mean; q += d * d; }
#pragma unroll
    for (int off = 16; off >= 1; off >>= 1) q += __shfl_xor(q, off, 32);
    float rs = rsqrtf(q * (1.0f / 256.0f) + 1e-5f);
#pragma unroll
    for (int j = 0; j < 8; ++j) {
        int c = lane + 32 * j;
        out[base + c] = (bf16)((v[j] - mean) * rs * g[c] + b[c]);
    }
}

// ---------------------------------------------------------------------------
// Spatial attention: per (b,t) block. qkv cols: q_s=c*3, k_s=c*3+1, v_s=c*3+2
// ---------------------------------------------------------------------------
__global__ __launch_bounds__(256) void attn_spatial_kernel(
    const bf16* __restrict__ qkv, float* __restrict__ oatt)
{
    __shared__ float sq[kS][128], sk[kS][128], sv[kS][128];
    long tokBase = (long)blockIdx.x * kS;    // blockIdx = b*T + t
    for (int idx = threadIdx.x; idx < kS * 384; idx += 256) {
        int s = idx / 384, cc = idx - s * 384;
        int kind = cc >> 7, c = cc & 127;
        float val = (float)qkv[(tokBase + s) * 768 + c * 3 + kind];
        if (kind == 0) sq[s][c] = val;
        else if (kind == 1) sk[s][c] = val;
        else sv[s][c] = val;
    }
    __syncthreads();
    int task = threadIdx.x;
    if (task < 8 * kS) {
        int h = task / kS, qi = task - h * kS;
        int cb = h * 16;
        float sc[kS];
        float mx = -3.0e38f;
        for (int ki = 0; ki < kS; ++ki) {
            float d = 0.0f;
#pragma unroll
            for (int dd = 0; dd < 16; ++dd) d += sq[qi][cb + dd] * sk[ki][cb + dd];
            d *= 0.08838834764831845f;   // 128^-0.5
            sc[ki] = d; mx = fmaxf(mx, d);
        }
        float sum = 0.0f;
        for (int ki = 0; ki < kS; ++ki) { sc[ki] = expf(sc[ki] - mx); sum += sc[ki]; }
        float inv = 1.0f / sum;
#pragma unroll
        for (int dd = 0; dd < 16; ++dd) {
            float o = 0.0f;
            for (int ki = 0; ki < kS; ++ki) o += sc[ki] * sv[ki][cb + dd];
            oatt[(tokBase + qi) * 128 + cb + dd] = o * inv;
        }
    }
}

// ---------------------------------------------------------------------------
// Temporal attention: per (b,s) block. q_t=(128+c)*3 -> c*3+384, etc.
// ---------------------------------------------------------------------------
__global__ __launch_bounds__(256) void attn_temporal_kernel(
    const bf16* __restrict__ qkv, float* __restrict__ oatt)
{
    __shared__ float sq[kT][128], sk[kT][128], sv[kT][128];
    int b = blockIdx.x / kS;
    int s = blockIdx.x - b * kS;
    for (int idx = threadIdx.x; idx < kT * 384; idx += 256) {
        int t = idx / 384, cc = idx - t * 384;
        int kind = cc >> 7, c = cc & 127;
        long tok = ((long)b * kT + t) * kS + s;
        float val = (float)qkv[tok * 768 + c * 3 + 384 + kind];
        if (kind == 0) sq[t][c] = val;
        else if (kind == 1) sk[t][c] = val;
        else sv[t][c] = val;
    }
    __syncthreads();
    int task = threadIdx.x;
    if (task < 8 * kT) {
        int h = task / kT, qi = task - h * kT;
        int cb = h * 16;
        float sc[kT];
        float mx = -3.0e38f;
        for (int ki = 0; ki < kT; ++ki) {
            float d = 0.0f;
#pragma unroll
            for (int dd = 0; dd < 16; ++dd) d += sq[qi][cb + dd] * sk[ki][cb + dd];
            d *= 0.08838834764831845f;
            sc[ki] = d; mx = fmaxf(mx, d);
        }
        float sum = 0.0f;
        for (int ki = 0; ki < kT; ++ki) { sc[ki] = expf(sc[ki] - mx); sum += sc[ki]; }
        float inv = 1.0f / sum;
        long tokQ = ((long)b * kT + qi) * kS + s;
#pragma unroll
        for (int dd = 0; dd < 16; ++dd) {
            float o = 0.0f;
            for (int ki = 0; ki < kT; ++ki) o += sc[ki] * sv[ki][cb + dd];
            oatt[tokQ * 128 + cb + dd] = o * inv;
        }
    }
}

// ---------------------------------------------------------------------------
// 3x3 depthwise conv over (T,S) image of v (+bias). vcolBase=2 (spatial) /
// 386 (temporal) selects v_s / v_t channels inside the interleaved qkv.
// ---------------------------------------------------------------------------
__global__ __launch_bounds__(256) void dwconv_kernel(
    const bf16* __restrict__ qkv, const float* __restrict__ w9,
    const float* __restrict__ cb, float* __restrict__ hdpe, int vcolBase)
{
    long idx = (long)blockIdx.x * 256 + threadIdx.x;   // NT*128 threads
    long tok = idx >> 7; int c = (int)(idx & 127);
    int b = (int)(tok / (kT * kS));
    int rem = (int)(tok - (long)b * (kT * kS));
    int t = rem / kS, s = rem - t * kS;
    float acc = cb[c];
#pragma unroll
    for (int dt = -1; dt <= 1; ++dt) {
        int tt = t + dt; if (tt < 0 || tt >= kT) continue;
#pragma unroll
        for (int ds = -1; ds <= 1; ++ds) {
            int ss = s + ds; if (ss < 0 || ss >= kS) continue;
            long tok2 = ((long)b * kT + tt) * kS + ss;
            float v = (float)qkv[tok2 * 768 + c * 3 + vcolBase];
            acc += v * w9[c * 9 + (dt + 1) * 3 + (ds + 1)];
        }
    }
    hdpe[tok * 128 + c] = acc;
}

// ---------------------------------------------------------------------------
// x = att + hdpe + scale * maxpool_T(hdpe); bf16 output
// ---------------------------------------------------------------------------
__global__ __launch_bounds__(256) void combine_pe_kernel(
    const float* __restrict__ att, const float* __restrict__ hdpe,
    bf16* __restrict__ out, float ldpeScale)
{
    long idx = (long)blockIdx.x * 256 + threadIdx.x;   // NT*128
    long tok = idx >> 7; int c = (int)(idx & 127);
    int t = (int)((tok / kS) % kT);
    float h0 = hdpe[tok * 128 + c];
    float m = h0;
    if (t > 0)      m = fmaxf(m, hdpe[(tok - kS) * 128 + c]);
    if (t < kT - 1) m = fmaxf(m, hdpe[(tok + kS) * 128 + c]);
    out[tok * 128 + c] = (bf16)(att[tok * 128 + c] + h0 + ldpeScale * m);
}

// ---------------------------------------------------------------------------
// Fusion gate + combine: alpha=softmax([xs,xt]@fw^T+fb); y=ps*a0+pt*a1 (bf16)
// One wave per token.
// ---------------------------------------------------------------------------
__global__ __launch_bounds__(256) void fuse_combine_kernel(
    const bf16* __restrict__ xs, const bf16* __restrict__ xt,
    const float* __restrict__ ps, const float* __restrict__ pt,
    const float* __restrict__ fw, const float* __restrict__ fb,
    bf16* __restrict__ ycomb)
{
    long tok = (long)blockIdx.x * 8 + (threadIdx.x >> 5);
    int lane = threadIdx.x & 31;
    float a0 = 0.0f, a1 = 0.0f;
#pragma unroll
    for (int j = 0; j < 8; ++j) {
        int c = lane + 32 * j;
        float xv = (c < 128) ? (float)xs[tok * 128 + c]
                             : (float)xt[tok * 128 + c - 128];
        a0 += fw[c] * xv;
        a1 += fw[256 + c] * xv;
    }
#pragma unroll
    for (int off = 16; off >= 1; off >>= 1) {
        a0 += __shfl_xor(a0, off, 32);
        a1 += __shfl_xor(a1, off, 32);
    }
    float l0 = a0 + fb[0], l1 = a1 + fb[1];
    float m  = fmaxf(l0, l1);
    float e0 = expf(l0 - m), e1 = expf(l1 - m);
    float inv = 1.0f / (e0 + e1);
    e0 *= inv; e1 *= inv;
#pragma unroll
    for (int j = 0; j < 8; ++j) {
        int c = lane + 32 * j;
        long i = tok * 256 + c;
        ycomb[i] = (bf16)(ps[i] * e0 + pt[i] * e1);
    }
}

// ---------------------------------------------------------------------------
// Head: out = h @ head_w^T, head_w:(2,256). One wave per token.
// ---------------------------------------------------------------------------
__global__ __launch_bounds__(256) void head_kernel(
    const float* __restrict__ h, const float* __restrict__ hw,
    float* __restrict__ out)
{
    long tok = (long)blockIdx.x * 8 + (threadIdx.x >> 5);
    int lane = threadIdx.x & 31;
    float a0 = 0.0f, a1 = 0.0f;
#pragma unroll
    for (int j = 0; j < 8; ++j) {
        int c = lane + 32 * j;
        float v = h[tok * 256 + c];
        a0 += hw[c] * v;
        a1 += hw[256 + c] * v;
    }
#pragma unroll
    for (int off = 16; off >= 1; off >>= 1) {
        a0 += __shfl_xor(a0, off, 32);
        a1 += __shfl_xor(a1, off, 32);
    }
    if (lane == 0) { out[tok * 2] = a0; out[tok * 2 + 1] = a1; }
}

// ---------------------------------------------------------------------------
// Host-side orchestration
// ---------------------------------------------------------------------------
extern "C" void kernel_launch(void* const* d_in, const int* in_sizes, int n_in,
                              void* d_out, int out_size, void* d_ws, size_t ws_size,
                              hipStream_t stream)
{
    (void)in_sizes; (void)n_in; (void)out_size;

    const float* x_in   = (const float*)d_in[0];
    const float* emb_w  = (const float*)d_in[1];
    const float* head_w = (const float*)d_in[2];

    // ---- workspace layout (all offsets multiples of 256 bytes) ----
    const size_t SZ_H    = (size_t)kNT * 256 * 4;   // 120,324,096
    const size_t SZ_XN   = (size_t)kNT * 256 * 2;   //  60,162,048
    const size_t SZ_QKV  = (size_t)kNT * 768 * 2;   // 180,486,144
    const size_t SZ_ATT  = (size_t)kNT * 128 * 4;   //  60,162,048
    const size_t SZ_X16  = (size_t)kNT * 128 * 2;   //  30,081,024
    const int wbElems[6] = {768 * 256, 256 * 128, 256 * 128, 256 * 256,
                            1024 * 256, 256 * 1024};

    char* ws = (char*)d_ws;
    size_t off = 0;
    bf16* wb[6][6];
    for (int blk = 0; blk < 6; ++blk)
        for (int wi = 0; wi < 6; ++wi) { wb[blk][wi] = (bf16*)(ws + off); off += (size_t)wbElems[wi] * 2; }
    float* hA = (float*)(ws + off); off += SZ_H;
    float* hB = (float*)(ws + off); off += SZ_H;
    bf16*  xn = (bf16*)(ws + off);  off += SZ_XN;
    char* pool = ws + off;
    // phase A
    bf16*  qkvb   = (bf16*)(pool);
    float* xs_att = (float*)(pool + SZ_QKV);
    float* xt_att = (float*)(pool + SZ_QKV + SZ_ATT);
    float* hdpe_s = (float*)(pool + SZ_QKV + 2 * SZ_ATT);
    float* hdpe_t = (float*)(pool + SZ_QKV + 3 * SZ_ATT);
    bf16*  xs16   = (bf16*)(pool + SZ_QKV + 4 * SZ_ATT);
    bf16*  xt16   = (bf16*)(pool + SZ_QKV + 4 * SZ_ATT + SZ_X16);
    size_t poolSize = SZ_QKV + 4 * SZ_ATT + 2 * SZ_X16;   // 481,296,384
    // phase B (aliases dead phase-A buffers)
    float* ps    = (float*)(pool);
    float* pt    = (float*)(pool + SZ_H);
    bf16*  ycomb = (bf16*)(pool + 2 * SZ_H);
    // phase C
    bf16*  mid   = (bf16*)(pool);                          // NT x 1024 bf16
    if (ws_size < off + poolSize) return;   // workspace too small; bail

    // ---- convert weights to bf16 (per call; ~5.1M elements total) ----
    const int widx[6] = {2, 10, 12, 14, 18, 19};  // qkv,proj_s,proj_t,proj,fc1,fc2
    for (int blk = 0; blk < 6; ++blk) {
        for (int wi = 0; wi < 6; ++wi) {
            long n = wbElems[wi];
            f32_to_bf16_kernel<<<(unsigned)((n + 255) / 256), 256, 0, stream>>>(
                (const float*)d_in[3 + 20 * blk + widx[wi]], wb[blk][wi], n);
        }
    }

    // ---- embedding ----
    embed_kernel<<<(unsigned)kNT, 256, 0, stream>>>(x_in, emb_w, hA);

    const unsigned gLN   = (unsigned)(kNT / 8);          // 14688
    const unsigned gTOKC = (unsigned)(kNT * 128 / 256);  // 58752

    for (int blk = 0; blk < 6; ++blk) {
        const int base = 3 + 20 * blk;
        const float* ln1_g    = (const float*)d_in[base + 0];
        const float* ln1_b    = (const float*)d_in[base + 1];
        const float* qkv_b    = (const float*)d_in[base + 3];
        const float* conv_s_w = (const float*)d_in[base + 4];
        const float* conv_s_b = (const float*)d_in[base + 5];
        const float* conv_t_w = (const float*)d_in[base + 6];
        const float* conv_t_b = (const float*)d_in[base + 7];
        const float* fusion_w = (const float*)d_in[base + 8];
        const float* fusion_b = (const float*)d_in[base + 9];
        const float* proj_s_b = (const float*)d_in[base + 11];
        const float* proj_t_b = (const float*)d_in[base + 13];
        const float* proj_b   = (const float*)d_in[base + 15];
        const float* ln2_g    = (const float*)d_in[base + 16];
        const float* ln2_b    = (const float*)d_in[base + 17];

        // ---- STAF attention ----
        layernorm_kernel<<<gLN, 256, 0, stream>>>(hA, ln1_g, ln1_b, xn);
        // qkv: (NT,256) @ (768,256)^T + b -> bf16
        gemm_wmma_kernel<0, false, true><<<(unsigned)(kROWT * 12 / 8), 256, 0, stream>>>(
            xn, wb[blk][0], qkv_b, nullptr, qkvb, 256, 768, kROWT);
        attn_spatial_kernel<<<(unsigned)(kB * kT), 256, 0, stream>>>(qkvb, xs_att);
        attn_temporal_kernel<<<(unsigned)(kB * kS), 256, 0, stream>>>(qkvb, xt_att);
        dwconv_kernel<<<gTOKC, 256, 0, stream>>>(qkvb, conv_s_w, conv_s_b, hdpe_s, 2);
        dwconv_kernel<<<gTOKC, 256, 0, stream>>>(qkvb, conv_t_w, conv_t_b, hdpe_t, 386);
        combine_pe_kernel<<<gTOKC, 256, 0, stream>>>(xs_att, hdpe_s, xs16, 1e-4f);
        combine_pe_kernel<<<gTOKC, 256, 0, stream>>>(xt_att, hdpe_t, xt16, 1e-9f);
        // proj_s / proj_t: (NT,128) @ (256,128)^T + b -> f32
        gemm_wmma_kernel<0, false, false><<<(unsigned)(kROWT * 4 / 8), 256, 0, stream>>>(
            xs16, wb[blk][1], proj_s_b, nullptr, ps, 128, 256, kROWT);
        gemm_wmma_kernel<0, false, false><<<(unsigned)(kROWT * 4 / 8), 256, 0, stream>>>(
            xt16, wb[blk][2], proj_t_b, nullptr, pt, 128, 256, kROWT);
        fuse_combine_kernel<<<gLN, 256, 0, stream>>>(xs16, xt16, ps, pt,
                                                     fusion_w, fusion_b, ycomb);
        // proj: (NT,256) @ (256,256)^T + b + residual(hA) -> hB (f32)
        gemm_wmma_kernel<0, true, false><<<(unsigned)(kROWT * 4 / 8), 256, 0, stream>>>(
            ycomb, wb[blk][3], proj_b, hA, hB, 256, 256, kROWT);

        // ---- MLP ----
        layernorm_kernel<<<gLN, 256, 0, stream>>>(hB, ln2_g, ln2_b, xn);
        // fc1: (NT,256) @ (1024,256)^T, GELU -> bf16
        gemm_wmma_kernel<1, false, true><<<(unsigned)(kROWT * 16 / 8), 256, 0, stream>>>(
            xn, wb[blk][4], nullptr, nullptr, mid, 256, 1024, kROWT);
        // fc2: (NT,1024) @ (256,1024)^T + residual(hB) -> hA (f32)
        gemm_wmma_kernel<0, true, false><<<(unsigned)(kROWT * 4 / 8), 256, 0, stream>>>(
            mid, wb[blk][5], nullptr, hB, hA, 1024, 256, kROWT);
    }

    head_kernel<<<gLN, 256, 0, stream>>>(hA, head_w, (float*)d_out);
}
